// Qwen3VLVisualModel_12360915878758
// MI455X (gfx1250) — compile-verified, matrix operand order checked by reading, not run on hardware
//
#include <hip/hip_runtime.h>
#include <hip/hip_bf16.h>
#include <math.h>
#include <stdint.h>

// ---------------- model dims ----------------
#define SQ     2304          // S = 48*48 tokens
#define HDIM   1152          // hidden
#define NHEAD  16
#define HD     72            // head dim
#define INTER  4304
#define DEPTH  4
#define MERGED 4608          // H * 2 * 2
#define OUTD   2048
#define GRID   48

// ---------------- WMMA types ----------------
typedef __bf16  v16bf __attribute__((ext_vector_type(16)));
typedef __bf16  v8bf  __attribute__((ext_vector_type(8)));
typedef float   v8f   __attribute__((ext_vector_type(8)));

union Frag { v16bf v; v8bf h[2]; };

static __device__ inline v8f wmma_bf16(const Frag& a, const Frag& b, v8f c) {
    return __builtin_amdgcn_wmma_f32_16x16x32_bf16(false, a.v, false, b.v,
                                                   (short)0, c, false, false);
}

static __device__ inline float gelu_tanh(float x) {
    float x3 = x * x * x;
    return 0.5f * x * (1.f + tanhf(0.7978845608028654f * (x + 0.044715f * x3)));
}

// ---------------- CDNA5 async copy global->LDS ----------------
typedef int v4i __attribute__((vector_size(16)));
typedef __attribute__((address_space(1))) v4i as1_v4i;
typedef __attribute__((address_space(3))) v4i as3_v4i;

#if __has_builtin(__builtin_amdgcn_global_load_async_to_lds_b128)
#define HAS_ASYNC_LDS 1
#else
#define HAS_ASYNC_LDS 0
#endif

// copy 16B from global (per-lane address) into LDS (per-lane address)
static __device__ inline void cp16_g2l(const float* g, float* l) {
#if HAS_ASYNC_LDS
    __builtin_amdgcn_global_load_async_to_lds_b128(
        (as1_v4i*)(unsigned long long)g,
        (as3_v4i*)(unsigned)(unsigned long long)l,
        0, 0);
#else
    float4 v = *(const float4*)g;
    *(float4*)l = v;
#endif
}

static __device__ inline void wait_async0() {
#if HAS_ASYNC_LDS
#if __has_builtin(__builtin_amdgcn_s_wait_asynccnt)
    __builtin_amdgcn_s_wait_asynccnt(0);
#else
    asm volatile("s_wait_asynccnt 0x0" ::: "memory");
#endif
#endif
}

// =====================================================================
// Generic GEMM:  C[M,N] = epi(A[M,K] @ W[N,K]^T + bias) (+ residual R)
// bf16 WMMA, f32 accumulate. Block = 256 threads = 8 waves.
// Block tile 64x64, K-step 32. Double-buffered async global->LDS staging
// of f32 tiles; bf16 conversion at fragment build (co-executes with XDL).
// =====================================================================
#define BM 64
#define BN 64
#define BK 32
#define LDSF 36          // f32 per LDS row (144B: 16B-aligned chunks, skewed banks)
#define LDS_STRIDE 40    // bf16 LDS stride (attention P staging)

template<int EPI>   // 0 = none, 1 = gelu
__global__ __launch_bounds__(256)
void gemm_bf16_wmma(const float* __restrict__ A, int lda,
                    const float* __restrict__ W,           // [N,K] row-major
                    const float* __restrict__ bias,        // [N] or null
                    const float* __restrict__ R, int ldr,  // residual or null
                    float* __restrict__ C, int ldc,
                    int M, int N, int K)
{
    __shared__ float sA[2][BM * LDSF];
    __shared__ float sB[2][BN * LDSF];

    const int t    = threadIdx.x;
    const int lane = t & 31;
    const int wave = t >> 5;
    const int bm0  = blockIdx.y * BM;
    const int bn0  = blockIdx.x * BN;

    // staging mapping: thread t -> row t/4 (0..63), two 16B chunks (t%4) and (t%4)+4
    const int lrow = t >> 2;
    const int c0   = (t & 3);        // chunk 0..3  (cols 0..15)
    const int c1   = c0 + 4;         // chunk 4..7  (cols 16..31)
    // clamp rows: garbage rows/cols are discarded by the epilogue guards
    const int gmA = (bm0 + lrow < M) ? bm0 + lrow : M - 1;
    const int gnW = (bn0 + lrow < N) ? bn0 + lrow : N - 1;

    // compute mapping
    const int wm = wave >> 1;          // 0..3  (16-row tile)
    const int wn = (wave & 1) * 2;     // 0 or 2 (two 16-col tiles)
    const int lr = lane & 15;
    const int kb = (lane >> 4) * 8;

    v8f acc0 = {0.f,0.f,0.f,0.f,0.f,0.f,0.f,0.f};
    v8f acc1 = {0.f,0.f,0.f,0.f,0.f,0.f,0.f,0.f};

    const int Kfull = K & ~(BK - 1);
    const int ns    = Kfull / BK;

    // prologue: stage 0
    if (ns > 0) {
        cp16_g2l(A + (size_t)gmA * lda + c0 * 4, &sA[0][lrow * LDSF + c0 * 4]);
        cp16_g2l(A + (size_t)gmA * lda + c1 * 4, &sA[0][lrow * LDSF + c1 * 4]);
        cp16_g2l(W + (size_t)gnW * K   + c0 * 4, &sB[0][lrow * LDSF + c0 * 4]);
        cp16_g2l(W + (size_t)gnW * K   + c1 * 4, &sB[0][lrow * LDSF + c1 * 4]);
    }

    for (int s = 0; s < ns; ++s) {
        wait_async0();          // my stage-s copies have landed
        __syncthreads();        // everyone's landed; prev compute done
        if (s + 1 < ns) {       // kick stage s+1 into the other buffer
            const int nb = (s + 1) & 1;
            const int k1 = (s + 1) * BK;
            cp16_g2l(A + (size_t)gmA * lda + k1 + c0 * 4, &sA[nb][lrow * LDSF + c0 * 4]);
            cp16_g2l(A + (size_t)gmA * lda + k1 + c1 * 4, &sA[nb][lrow * LDSF + c1 * 4]);
            cp16_g2l(W + (size_t)gnW * K   + k1 + c0 * 4, &sB[nb][lrow * LDSF + c0 * 4]);
            cp16_g2l(W + (size_t)gnW * K   + k1 + c1 * 4, &sB[nb][lrow * LDSF + c1 * 4]);
        }
        // ---- compute stage s (overlaps stage s+1 copies) ----
        const int b = s & 1;
        Frag fa, fb0, fb1;
        {
            const float* pa = &sA[b][(wm * 16 + lr) * LDSF + kb];
            const float* p0 = &sB[b][((wn + 0) * 16 + lr) * LDSF + kb];
            const float* p1 = &sB[b][((wn + 1) * 16 + lr) * LDSF + kb];
            #pragma unroll
            for (int e = 0; e < 8; ++e) {
                fa.v[e]      = (__bf16)pa[e];   fa.v[8 + e]  = (__bf16)pa[16 + e];
                fb0.v[e]     = (__bf16)p0[e];   fb0.v[8 + e] = (__bf16)p0[16 + e];
                fb1.v[e]     = (__bf16)p1[e];   fb1.v[8 + e] = (__bf16)p1[16 + e];
            }
        }
        acc0 = wmma_bf16(fa, fb0, acc0);
        acc1 = wmma_bf16(fa, fb1, acc1);
    }

    // ---- K tail (K % 32 != 0, e.g. K=4304): guarded register staging ----
    if (Kfull < K) {
        __syncthreads();
        {
            const float* ap = A + (size_t)gmA * lda;
            const float* wp = W + (size_t)gnW * K;
            #pragma unroll
            for (int e = 0; e < 4; ++e) {
                int ka = Kfull + c0 * 4 + e;
                int kbc = Kfull + c1 * 4 + e;
                sA[0][lrow * LDSF + c0 * 4 + e] = (ka  < K) ? ap[ka]  : 0.f;
                sA[0][lrow * LDSF + c1 * 4 + e] = (kbc < K) ? ap[kbc] : 0.f;
                sB[0][lrow * LDSF + c0 * 4 + e] = (ka  < K) ? wp[ka]  : 0.f;
                sB[0][lrow * LDSF + c1 * 4 + e] = (kbc < K) ? wp[kbc] : 0.f;
            }
        }
        __syncthreads();
        Frag fa, fb0, fb1;
        {
            const float* pa = &sA[0][(wm * 16 + lr) * LDSF + kb];
            const float* p0 = &sB[0][((wn + 0) * 16 + lr) * LDSF + kb];
            const float* p1 = &sB[0][((wn + 1) * 16 + lr) * LDSF + kb];
            #pragma unroll
            for (int e = 0; e < 8; ++e) {
                fa.v[e]      = (__bf16)pa[e];   fa.v[8 + e]  = (__bf16)pa[16 + e];
                fb0.v[e]     = (__bf16)p0[e];   fb0.v[8 + e] = (__bf16)p0[16 + e];
                fb1.v[e]     = (__bf16)p1[e];   fb1.v[8 + e] = (__bf16)p1[16 + e];
            }
        }
        acc0 = wmma_bf16(fa, fb0, acc0);
        acc1 = wmma_bf16(fa, fb1, acc1);
    }

    // ---- epilogue ----
    #pragma unroll
    for (int j = 0; j < 8; ++j) {
        int m  = wm * 16 + j + ((lane >> 4) << 3);
        int gm = bm0 + m;
        #pragma unroll
        for (int tb = 0; tb < 2; ++tb) {
            int n  = (wn + tb) * 16 + lr;
            int gn = bn0 + n;
            if (gm < M && gn < N) {
                float v = (tb == 0) ? acc0[j] : acc1[j];
                if (bias) v += bias[gn];
                if (EPI == 1) v = gelu_tanh(v);
                if (R) v += R[(size_t)gm * ldr + gn];
                C[(size_t)gm * ldc + gn] = v;
            }
        }
    }
}

// =====================================================================
// LayerNorm: one block per row of width Hw
// =====================================================================
__global__ __launch_bounds__(256)
void ln_kernel(const float* __restrict__ x, const float* __restrict__ w,
               const float* __restrict__ b, float* __restrict__ y, int Hw)
{
    __shared__ float r1[256];
    __shared__ float r2[256];
    const int row = blockIdx.x;
    const int tid = threadIdx.x;
    const float* xr = x + (size_t)row * Hw;

    float s = 0.f, s2 = 0.f;
    for (int j = tid; j < Hw; j += 256) { float v = xr[j]; s += v; s2 += v * v; }
    r1[tid] = s; r2[tid] = s2;
    __syncthreads();
    for (int off = 128; off > 0; off >>= 1) {
        if (tid < off) { r1[tid] += r1[tid + off]; r2[tid] += r2[tid + off]; }
        __syncthreads();
    }
    float mean = r1[0] / (float)Hw;
    float var  = r2[0] / (float)Hw - mean * mean;
    float rstd = rsqrtf(var + 1e-6f);
    float* yr = y + (size_t)row * Hw;
    for (int j = tid; j < Hw; j += 256)
        yr[j] = w[j] * ((xr[j] - mean) * rstd) + b[j];
}

// =====================================================================
// pos-id helper (arange(2304).reshape(24,2,24,2).transpose(0,2,1,3) flatten)
// =====================================================================
static __device__ inline int pos_id_of(int s) {
    int p  = s / 96;
    int r  = (s % 96) / 4;
    int q  = (s % 4) / 2;
    int e  = s & 1;
    return p * 96 + q * 48 + r * 2 + e;
}

__global__ void pos_add_kernel(float* __restrict__ x, const float* __restrict__ pos_tab)
{
    int idx = blockIdx.x * blockDim.x + threadIdx.x;
    if (idx >= SQ * HDIM) return;
    int s = idx / HDIM, j = idx % HDIM;
    x[idx] += pos_tab[(size_t)pos_id_of(s) * HDIM + j];
}

// =====================================================================
// RoPE on q,k inside qkv [S, 3, NH, HD]
// =====================================================================
__global__ void rope_kernel(float* __restrict__ qkv)
{
    int idx = blockIdx.x * blockDim.x + threadIdx.x;
    const int total = SQ * NHEAD * 36;
    if (idx >= total) return;
    int s  = idx / (NHEAD * 36);
    int r  = idx % (NHEAD * 36);
    int hh = r / 36;
    int d  = r % 36;

    int pid = pos_id_of(s);
    float pos = (d < 18) ? (float)(pid / GRID) : (float)(pid % GRID);
    int   fi  = (d < 18) ? d : d - 18;
    float invf = __powf(10000.f, -(2.f * (float)fi) / 36.f);
    float ang = pos * invf;
    float c = __cosf(ang), sn = __sinf(ang);

    float* q = qkv + ((size_t)(s * 3 + 0) * NHEAD + hh) * HD;
    float* k = qkv + ((size_t)(s * 3 + 1) * NHEAD + hh) * HD;
    float q1 = q[d], q2 = q[d + 36];
    q[d]      = q1 * c - q2 * sn;
    q[d + 36] = q2 * c + q1 * sn;
    float k1 = k[d], k2 = k[d + 36];
    k[d]      = k1 * c - k2 * sn;
    k[d + 36] = k2 * c + k1 * sn;
}

// =====================================================================
// Flash attention, one wave per (head, 16-row q tile).
// Single segment (cu_seqlens=[0,S]) -> no mask. WMMA for QK^T and P V.
// =====================================================================
__global__ __launch_bounds__(256)
void attn_kernel(const float* __restrict__ qkv, float* __restrict__ o)
{
    __shared__ __bf16 sP[8 * 16 * LDS_STRIDE];   // per-wave P staging

    const int t    = threadIdx.x;
    const int lane = t & 31;
    const int wave = t >> 5;
    const int gw   = blockIdx.x * 8 + wave;
    const int head = gw % NHEAD;
    const int qt   = gw / NHEAD;          // 0..143
    __bf16* sp = &sP[wave * 16 * LDS_STRIDE];

    const int lr = lane & 15;
    const int kb = (lane >> 4) * 8;
    const float scale = 0.11785113019775793f;   // 1/sqrt(72)

    // ---- Q fragments (16 x 96, d zero-padded past 72), scale folded in ----
    Frag fq[3];
    {
        int qrow = qt * 16 + lr;
        const float* qp = qkv + ((size_t)(qrow * 3 + 0) * NHEAD + head) * HD;
        #pragma unroll
        for (int ds = 0; ds < 3; ++ds) {
            #pragma unroll
            for (int e = 0; e < 8; ++e) {
                int d0 = ds * 32 + kb + e;
                int d1 = d0 + 16;
                fq[ds].v[e]     = (__bf16)((d0 < HD) ? qp[d0] * scale : 0.f);
                fq[ds].v[8 + e] = (__bf16)((d1 < HD) ? qp[d1] * scale : 0.f);
            }
        }
    }

    v8f accO[5];
    #pragma unroll
    for (int i = 0; i < 5; ++i) accO[i] = (v8f){0.f,0.f,0.f,0.f,0.f,0.f,0.f,0.f};
    float rmax[8], rsum[8];
    #pragma unroll
    for (int j = 0; j < 8; ++j) { rmax[j] = -1e30f; rsum[j] = 0.f; }

    for (int kv = 0; kv < SQ; kv += 32) {
        // ---- scores tile 16x32 ----
        v8f s0 = (v8f){0.f,0.f,0.f,0.f,0.f,0.f,0.f,0.f};
        v8f s1 = (v8f){0.f,0.f,0.f,0.f,0.f,0.f,0.f,0.f};
        #pragma unroll
        for (int nt = 0; nt < 2; ++nt) {
            int kid = kv + nt * 16 + lr;
            const float* kp = qkv + ((size_t)(kid * 3 + 1) * NHEAD + head) * HD;
            #pragma unroll
            for (int ds = 0; ds < 3; ++ds) {
                Frag fk;
                #pragma unroll
                for (int e = 0; e < 8; ++e) {
                    int d0 = ds * 32 + kb + e;
                    int d1 = d0 + 16;
                    fk.v[e]     = (__bf16)((d0 < HD) ? kp[d0] : 0.f);
                    fk.v[8 + e] = (__bf16)((d1 < HD) ? kp[d1] : 0.f);
                }
                if (nt == 0) s0 = wmma_bf16(fq[ds], fk, s0);
                else         s1 = wmma_bf16(fq[ds], fk, s1);
            }
        }

        // ---- online softmax, write P to LDS ----
        #pragma unroll
        for (int j = 0; j < 8; ++j) {
            float a = s0[j], b = s1[j];
            float mloc = fmaxf(a, b);
            #pragma unroll
            for (int off = 1; off < 16; off <<= 1)
                mloc = fmaxf(mloc, __shfl_xor(mloc, off, 32));
            float mnew  = fmaxf(rmax[j], mloc);
            float alpha = __expf(rmax[j] - mnew);
            rmax[j] = mnew;
            rsum[j] *= alpha;
            #pragma unroll
            for (int i = 0; i < 5; ++i) accO[i][j] *= alpha;
            float p0 = __expf(a - mnew);
            float p1 = __expf(b - mnew);
            float ps = p0 + p1;
            #pragma unroll
            for (int off = 1; off < 16; off <<= 1)
                ps += __shfl_xor(ps, off, 32);
            rsum[j] += ps;
            int prow = j + ((lane >> 4) << 3);
            sp[prow * LDS_STRIDE + lr]      = (__bf16)p0;
            sp[prow * LDS_STRIDE + 16 + lr] = (__bf16)p1;
        }
        __builtin_amdgcn_wave_barrier();   // same-wave LDS in-order; block reordering

        // ---- P fragment (A layout) from LDS ----
        Frag fp;
        {
            const __bf16* base = &sp[lr * LDS_STRIDE + kb];
            fp.h[0] = *(const v8bf*)(base);
            fp.h[1] = *(const v8bf*)(base + 16);
        }

        // ---- O += P @ V  (V: 32 x 80 padded) ----
        #pragma unroll
        for (int ot = 0; ot < 5; ++ot) {
            Frag fv;
            int d = ot * 16 + lr;
            bool dv = d < HD;
            #pragma unroll
            for (int e = 0; e < 8; ++e) {
                int k0i = kv + kb + e;
                int k1i = kv + kb + 16 + e;
                fv.v[e]     = (__bf16)(dv ? qkv[((size_t)(k0i * 3 + 2) * NHEAD + head) * HD + d] : 0.f);
                fv.v[8 + e] = (__bf16)(dv ? qkv[((size_t)(k1i * 3 + 2) * NHEAD + head) * HD + d] : 0.f);
            }
            accO[ot] = wmma_bf16(fp, fv, accO[ot]);
        }
    }

    // ---- normalize + store ----
    #pragma unroll
    for (int j = 0; j < 8; ++j) {
        float inv = 1.f / rsum[j];
        int row = qt * 16 + j + ((lane >> 4) << 3);
        #pragma unroll
        for (int ot = 0; ot < 5; ++ot) {
            int d = ot * 16 + lr;
            if (d < HD)
                o[(size_t)row * HDIM + head * HD + d] = accO[ot][j] * inv;
        }
    }
}

// =====================================================================
// host side
// =====================================================================
static inline unsigned cdiv(unsigned a, unsigned b) { return (a + b - 1) / b; }

extern "C" void kernel_launch(void* const* d_in, const int* in_sizes, int n_in,
                              void* d_out, int out_size, void* d_ws, size_t ws_size,
                              hipStream_t stream)
{
    const float* pixel   = (const float*)d_in[0];
    const float* conv_w  = (const float*)d_in[2];
    const float* conv_b  = (const float*)d_in[3];
    const float* pos_tab = (const float*)d_in[4];
    const float* ln1_w   = (const float*)d_in[5];
    const float* ln1_b   = (const float*)d_in[6];
    const float* ln2_w   = (const float*)d_in[7];
    const float* ln2_b   = (const float*)d_in[8];
    const float* qkv_w   = (const float*)d_in[9];
    const float* qkv_b   = (const float*)d_in[10];
    const float* proj_w  = (const float*)d_in[11];
    const float* proj_b  = (const float*)d_in[12];
    const float* fc1_w   = (const float*)d_in[13];
    const float* fc1_b   = (const float*)d_in[14];
    const float* fc2_w   = (const float*)d_in[15];
    const float* fc2_b   = (const float*)d_in[16];
    const float* mn_w    = (const float*)d_in[17];
    const float* mn_b    = (const float*)d_in[18];
    const float* mfc1_w  = (const float*)d_in[19];
    const float* mfc1_b  = (const float*)d_in[20];
    const float* mfc2_w  = (const float*)d_in[21];
    const float* mfc2_b  = (const float*)d_in[22];
    float* out = (float*)d_out;

    float* x     = (float*)d_ws;
    float* h     = x    + (size_t)SQ * HDIM;
    float* qkv   = h    + (size_t)SQ * HDIM;
    float* obuf  = qkv  + (size_t)SQ * 3 * HDIM;
    float* inter = obuf + (size_t)SQ * HDIM;
    float* mg1   = inter + (size_t)SQ * INTER;

    const int PKK = 1536;  // patch K = C*T*P*P

    // 1) patch embed: x = pixel @ conv_w^T + conv_b
    gemm_bf16_wmma<0><<<dim3(cdiv(HDIM, BN), cdiv(SQ, BM)), 256, 0, stream>>>(
        pixel, PKK, conv_w, conv_b, nullptr, 0, x, HDIM, SQ, HDIM, PKK);
    // 2) + pos_tab[pos_ids]
    pos_add_kernel<<<cdiv(SQ * HDIM, 256), 256, 0, stream>>>(x, pos_tab);

    for (int i = 0; i < DEPTH; ++i) {
        // LN1
        ln_kernel<<<SQ, 256, 0, stream>>>(x, ln1_w + (size_t)i * HDIM, ln1_b + (size_t)i * HDIM, h, HDIM);
        // QKV
        gemm_bf16_wmma<0><<<dim3(cdiv(3 * HDIM, BN), cdiv(SQ, BM)), 256, 0, stream>>>(
            h, HDIM, qkv_w + (size_t)i * 3 * HDIM * HDIM, qkv_b + (size_t)i * 3 * HDIM,
            nullptr, 0, qkv, 3 * HDIM, SQ, 3 * HDIM, HDIM);
        // RoPE
        rope_kernel<<<cdiv(SQ * NHEAD * 36, 256), 256, 0, stream>>>(qkv);
        // attention
        attn_kernel<<<(SQ / 16) * NHEAD / 8, 256, 0, stream>>>(qkv, obuf);
        // proj + residual
        gemm_bf16_wmma<0><<<dim3(cdiv(HDIM, BN), cdiv(SQ, BM)), 256, 0, stream>>>(
            obuf, HDIM, proj_w + (size_t)i * HDIM * HDIM, proj_b + (size_t)i * HDIM,
            x, HDIM, x, HDIM, SQ, HDIM, HDIM);
        // LN2
        ln_kernel<<<SQ, 256, 0, stream>>>(x, ln2_w + (size_t)i * HDIM, ln2_b + (size_t)i * HDIM, h, HDIM);
        // fc1 + gelu
        gemm_bf16_wmma<1><<<dim3(cdiv(INTER, BN), cdiv(SQ, BM)), 256, 0, stream>>>(
            h, HDIM, fc1_w + (size_t)i * INTER * HDIM, fc1_b + (size_t)i * INTER,
            nullptr, 0, inter, INTER, SQ, INTER, HDIM);
        // fc2 + residual
        gemm_bf16_wmma<0><<<dim3(cdiv(HDIM, BN), cdiv(SQ, BM)), 256, 0, stream>>>(
            inter, INTER, fc2_w + (size_t)i * HDIM * INTER, fc2_b + (size_t)i * HDIM,
            x, HDIM, x, HDIM, SQ, HDIM, INTER);
    }

    // merger: LN over H, then view [576, 4608]
    ln_kernel<<<SQ, 256, 0, stream>>>(x, mn_w, mn_b, h, HDIM);
    const int MROWS = SQ / 4;  // 576
    gemm_bf16_wmma<1><<<dim3(cdiv(MERGED, BN), cdiv(MROWS, BM)), 256, 0, stream>>>(
        h, MERGED, mfc1_w, mfc1_b, nullptr, 0, mg1, MERGED, MROWS, MERGED, MERGED);
    gemm_bf16_wmma<0><<<dim3(cdiv(OUTD, BN), cdiv(MROWS, BM)), 256, 0, stream>>>(
        mg1, MERGED, mfc2_w, mfc2_b, nullptr, 0, out, OUTD, MROWS, OUTD, MERGED);

    (void)in_sizes; (void)n_in; (void)out_size; (void)ws_size;
}
